// GCN_25847113187633
// MI455X (gfx1250) — matrix operand order repaired and checked by reading, not compiled, not run on hardware
//
#include <hip/hip_runtime.h>

typedef __bf16 bf16_t;
typedef bf16_t v16bf __attribute__((ext_vector_type(16)));
typedef float  v8f   __attribute__((ext_vector_type(8)));

#define DIN 256

__device__ __forceinline__ bf16_t f2bf(float f) {
  unsigned u = __builtin_bit_cast(unsigned, f);
  unsigned r = (u + 0x7FFFu + ((u >> 16) & 1u)) >> 16;   // round-to-nearest-even
  return __builtin_bit_cast(bf16_t, (unsigned short)r);
}

__device__ __forceinline__ float gelu_tanh(float x) {
  float x3 = x * x * x;
  float t = tanhf(0.7978845608028654f * (x + 0.044715f * x3));
  return 0.5f * x * (1.0f + t);
}

// ---------------- degree / norm ----------------
__global__ void k_deg_init(float* deg, int n) {
  int i = blockIdx.x * blockDim.x + threadIdx.x;
  if (i < n) deg[i] = 1.0f;                      // self-loop contribution
}

__global__ void k_deg_edges(const int* __restrict__ row, float* deg, int e) {
  int i = blockIdx.x * blockDim.x + threadIdx.x;
  if (i < e) atomicAdd(&deg[row[i]], 1.0f);
}

__global__ void k_rsqrt_inplace(float* deg, int n) {
  int i = blockIdx.x * blockDim.x + threadIdx.x;
  if (i < n) deg[i] = rsqrtf(deg[i]);
}

// ---------------- weight convert f32 -> bf16 ----------------
__global__ void k_w2bf(const float* __restrict__ w, bf16_t* __restrict__ o, int n) {
  int i = blockIdx.x * blockDim.x + threadIdx.x;
  if (i < n) o[i] = f2bf(w[i]);
}

// ---------------- Hs = dinv (x) H ; write to both A (gather src) and B (accumulator init)
__global__ void k_prescale(const float* __restrict__ src, const float* __restrict__ dinv,
                           float* __restrict__ A, float* __restrict__ B, int total) {
  int idx = blockIdx.x * blockDim.x + threadIdx.x;
  if (idx < total) {
    float v = src[idx] * dinv[idx >> 8];        // row = idx / 256
    A[idx] = v;
    B[idx] = v;
  }
}

// ---------------- edge scatter: out[row] += Hs[col], one wave32 per edge ----------------
__global__ void k_scatter(const float* __restrict__ Hs, const int* __restrict__ row,
                          const int* __restrict__ col, float* __restrict__ out, int e) {
  int wave = threadIdx.x >> 5;
  int lane = threadIdx.x & 31;
  int eidx = blockIdx.x * (blockDim.x >> 5) + wave;
  if (eidx >= e) return;
  int r = row[eidx], c = col[eidx];
  const float* s = Hs + (long)c * DIN;
  float*       d = out + (long)r * DIN;
#pragma unroll
  for (int i = 0; i < 8; ++i) {
    int dc = lane + i * 32;                     // coalesced 32-lane dword bursts
    atomicAdd(&d[dc], s[dc]);
  }
}

// ---------------- fused (dinv ⊙ B) @ W^T + bias [+ GELU], WMMA bf16 ----------------
// blockDim = 128 (4 waves). One block = 16-row strip x DOUT cols.
template <int DOUT, bool ACT>
__global__ __launch_bounds__(128) void k_gemm(const float* __restrict__ Hin,
                                              const bf16_t* __restrict__ Wb,
                                              const float* __restrict__ bias,
                                              const float* __restrict__ rowscale,
                                              float* __restrict__ Out, int nrows) {
  __shared__ bf16_t sA[16][DIN + 8];
  const int row0 = blockIdx.x * 16;
  const int tid = threadIdx.x;

  // stage 16x256 tile: f32 -> (scale by dinv) -> bf16 in LDS
  {
    int r  = tid >> 3;                // 0..15
    int c0 = (tid & 7) * 32;          // 0..224
    int rr = row0 + r;
    if (rr >= nrows) rr = nrows - 1;
    float sc = rowscale[rr];
    const float* src = Hin + (long)rr * DIN + c0;
#pragma unroll
    for (int j = 0; j < 32; ++j) sA[r][c0 + j] = f2bf(src[j] * sc);
  }
  __syncthreads();

  const int wave = tid >> 5;
  const int lane = tid & 31;
  const int half = lane >> 4;         // 0|1
  const int l15  = lane & 15;
  constexpr int NT = DOUT / 64;       // 16-col tiles per wave (4 waves cover DOUT)

  v8f acc[NT] = {};

#pragma unroll
  for (int ks = 0; ks < DIN / 32; ++ks) {
    const int k0 = ks * 32;
    // A fragment (16x32 bf16): lane row=l15; K = half*8+j and 16+half*8+j
    v16bf a;
#pragma unroll
    for (int j = 0; j < 8; ++j) {
      a[j]     = sA[l15][k0 + half * 8 + j];
      a[8 + j] = sA[l15][k0 + 16 + half * 8 + j];
    }
#pragma unroll
    for (int t = 0; t < NT; ++t) {
      const int col0 = (wave * NT + t) * 16;
      // B fragment (32x16): lane col=col0+l15; K = half*8+j and 16+half*8+j
      const bf16_t* wrow = Wb + (long)(col0 + l15) * DIN + k0 + half * 8;
      v16bf b;
#pragma unroll
      for (int j = 0; j < 8; ++j) { b[j] = wrow[j]; b[8 + j] = wrow[16 + j]; }
      acc[t] = __builtin_amdgcn_wmma_f32_16x16x32_bf16(false, a, false, b,
                                                       (short)0, acc[t], false, false);
    }
  }

  // epilogue: C/D layout -> lanes 0-15: M=i, lanes 16-31: M=8+i
  const int mbase = half * 8;
#pragma unroll
  for (int t = 0; t < NT; ++t) {
    const int c  = (wave * NT + t) * 16 + l15;
    const float bv = bias[c];
#pragma unroll
    for (int i = 0; i < 8; ++i) {
      int m = row0 + mbase + i;
      if (m < nrows) {
        float v = acc[t][i] + bv;
        if (ACT) v = gelu_tanh(v);
        Out[(long)m * DOUT + c] = v;
      }
    }
  }
}

extern "C" void kernel_launch(void* const* d_in, const int* in_sizes, int n_in,
                              void* d_out, int out_size, void* d_ws, size_t ws_size,
                              hipStream_t stream) {
  const float* X  = (const float*)d_in[0];
  const int*   row = (const int*)d_in[1];
  const int*   col = (const int*)d_in[2];
  const float* W0 = (const float*)d_in[3];
  const float* b0 = (const float*)d_in[4];
  const float* W1 = (const float*)d_in[5];
  const float* b1 = (const float*)d_in[6];
  const float* W2 = (const float*)d_in[7];
  const float* b2 = (const float*)d_in[8];

  const int N = in_sizes[0] / DIN;     // 100000
  const int E = in_sizes[1];           // 1600000
  const int total = N * DIN;

  // carve workspace (256B aligned): dinv | W0b | W1b | W2b | A | B
  char* ws = (char*)d_ws;
  size_t off = 0;
  auto carve = [&](size_t bytes) -> char* {
    char* p = ws + off;
    off += (bytes + 255) & ~(size_t)255;
    return p;
  };
  float*  dinv = (float*)carve((size_t)N * 4);
  bf16_t* W0b  = (bf16_t*)carve((size_t)256 * 256 * 2);
  bf16_t* W1b  = (bf16_t*)carve((size_t)256 * 256 * 2);
  bf16_t* W2b  = (bf16_t*)carve((size_t)128 * 256 * 2);
  float*  A    = (float*)carve((size_t)total * 4);
  float*  B    = (float*)carve((size_t)total * 4);
  (void)ws_size; (void)n_in;

  const int TB = 256;
  // normalization vector
  k_deg_init<<<(N + TB - 1) / TB, TB, 0, stream>>>(dinv, N);
  k_deg_edges<<<(E + TB - 1) / TB, TB, 0, stream>>>(row, dinv, E);
  k_rsqrt_inplace<<<(N + TB - 1) / TB, TB, 0, stream>>>(dinv, N);
  // bf16 weights
  k_w2bf<<<(256 * 256 + TB - 1) / TB, TB, 0, stream>>>(W0, W0b, 256 * 256);
  k_w2bf<<<(256 * 256 + TB - 1) / TB, TB, 0, stream>>>(W1, W1b, 256 * 256);
  k_w2bf<<<(128 * 256 + TB - 1) / TB, TB, 0, stream>>>(W2, W2b, 128 * 256);

  const int gridE = (E + 7) / 8;             // 8 waves of 32 per block, 1 edge per wave
  const int gridG = (N + 15) / 16;           // 6250 row strips
  const int gridP = (total + TB - 1) / TB;

  // layer 1: H = gelu(spmm(X) @ W0^T + b0)   (final dinv scale fused into GEMM)
  k_prescale<<<gridP, TB, 0, stream>>>(X, dinv, A, B, total);
  k_scatter<<<gridE, TB, 0, stream>>>(A, row, col, B, E);
  k_gemm<256, true><<<gridG, 128, 0, stream>>>(B, W0b, b0, dinv, A, N);

  // layer 2
  k_prescale<<<gridP, TB, 0, stream>>>(A, dinv, A, B, total);
  k_scatter<<<gridE, TB, 0, stream>>>(A, row, col, B, E);
  k_gemm<256, true><<<gridG, 128, 0, stream>>>(B, W1b, b1, dinv, A, N);

  // layer 3 (identity activation) -> d_out [N,128]
  k_prescale<<<gridP, TB, 0, stream>>>(A, dinv, A, B, total);
  k_scatter<<<gridE, TB, 0, stream>>>(A, row, col, B, E);
  k_gemm<128, false><<<gridG, 128, 0, stream>>>(B, W2b, b2, dinv, (float*)d_out, N);
}
